// TSAM_4664334483783
// MI455X (gfx1250) — matrix operand order repaired and checked by reading, not compiled, not run on hardware
//
#include <hip/hip_runtime.h>

// ---------------------------------------------------------------------------
// TSAM attention for MI455X (gfx1250, wave32, WMMA).
//   q = x.reshape(1024, 16384)
//   energy = q @ q^T                  (GEMM1: 1024x1024x16384, 4-way K-split)
//   attn   = softmax(rowmax - e)      == exp(rowmin - e) / sum
//   out    = gamma * (attn @ q) + x   (GEMM2: 1024x16384x1024, fused epilogue)
// bf16 operands / f32 accum via v_wmma_f32_16x16x32_bf16 (8x FLOP/instr vs the
// f32 WMMA path). All intermediates are L2-resident (192 MB); HBM traffic is
// ~read x + write out, so the kernels are WMMA-throughput bound. Per-wave
// tiles are 32x64 -> 8 WMMAs per 12 b128 loads per K-step; the K-loops are
// unrolled x2 so next-iteration loads overlap current WMMAs (partial
// s_wait_loadcnt drains instead of full serialization).
// ---------------------------------------------------------------------------

typedef __bf16 bf16;
typedef bf16  v16bf __attribute__((ext_vector_type(16)));
typedef bf16  v8bf  __attribute__((ext_vector_type(8)));
typedef float v8f   __attribute__((ext_vector_type(8)));

#define C_DIM  1024
#define HW     16384
#define KSPLIT 4

__device__ inline v8f wmma_bf16(v16bf a, v16bf b, v8f c) {
    return __builtin_amdgcn_wmma_f32_16x16x32_bf16(false, a, false, b,
                                                   (short)0, c, false, false);
}

// --- WMMA fragment loaders (layouts per CDNA5 ISA §7.12.2) -------
// A (16x32 bf16, MxK): lane 0-15 -> row M=lane, VGPR0..3 = K 0..7, VGPR4..7 =
// K 16..23; lanes 16-31 -> K 8..15 / 24..31. Two 16B loads per lane.
__device__ inline v16bf frag_a(const bf16* base, int ld, int row0, int k0, int lane) {
    int m  = lane & 15;
    int kb = (lane >> 4) << 3;                 // 0 or 8
    const bf16* p = base + (size_t)(row0 + m) * ld + (k0 + kb);
    union { v16bf v; v8bf h[2]; } u;
    u.h[0] = *(const v8bf*)(p);                // K = kb .. kb+7
    u.h[1] = *(const v8bf*)(p + 16);           // K = kb+16 .. kb+23
    return u.v;
}

// B (32x16 bf16, KxN): lane holds column N=lane&15; lanes 0-15 K=0..7,16..23,
// lanes 16-31 K=8..15,24..31. Operand storage is "n-major, k-contiguous":
// q row-major for GEMM1, qbT for GEMM2.
__device__ inline v16bf frag_b(const bf16* base, int ld, int n0, int k0, int lane) {
    int n  = lane & 15;
    int ks = (lane >> 4) << 3;                 // 0 or 8
    const bf16* p = base + (size_t)(n0 + n) * ld + (k0 + ks);
    union { v16bf v; v8bf h[2]; } u;
    u.h[0] = *(const v8bf*)(p);                // K = ks .. ks+7
    u.h[1] = *(const v8bf*)(p + 16);           // K = ks+16 .. ks+23
    return u.v;
}

// --- Kernel 1: fp32 -> bf16 pack + LDS-tiled transpose ----------------------
__global__ __launch_bounds__(256)
void tsam_pack(const float* __restrict__ x, bf16* __restrict__ qb,
               bf16* __restrict__ qbT) {
    __shared__ float tile[32][33];
    int tx = threadIdx.x, ty = threadIdx.y;
    int col0 = blockIdx.x * 32, row0 = blockIdx.y * 32;
    #pragma unroll
    for (int i = 0; i < 32; i += 8) {
        float v = x[(size_t)(row0 + ty + i) * HW + col0 + tx];
        qb[(size_t)(row0 + ty + i) * HW + col0 + tx] = (bf16)v;
        tile[ty + i][tx] = v;
    }
    __syncthreads();
    #pragma unroll
    for (int i = 0; i < 32; i += 8) {
        qbT[(size_t)(col0 + ty + i) * C_DIM + row0 + tx] = (bf16)tile[tx][ty + i];
    }
}

// --- Kernel 2: energy partials = qb @ qb^T (wave -> 32x64 block, K/4 slice) -
__global__ __launch_bounds__(32)
void tsam_gemm1(const bf16* __restrict__ qb, float* __restrict__ energyp) {
    int lane = threadIdx.x;
    int n0 = blockIdx.x * 64;
    int m0 = blockIdx.y * 32;
    int p  = blockIdx.z;                       // K-split slice
    const int KS = HW / KSPLIT;                // 4096
    int kbeg = p * KS, kend = kbeg + KS;

    v8f c0[4] = {{}, {}, {}, {}};
    v8f c1[4] = {{}, {}, {}, {}};
    #pragma unroll 2
    for (int k0 = kbeg; k0 < kend; k0 += 32) {
        v16bf a0 = frag_a(qb, HW, m0,      k0, lane);
        v16bf a1 = frag_a(qb, HW, m0 + 16, k0, lane);
        v16bf b[4];
        #pragma unroll
        for (int j = 0; j < 4; ++j) b[j] = frag_b(qb, HW, n0 + 16 * j, k0, lane);
        #pragma unroll
        for (int j = 0; j < 4; ++j) {
            c0[j] = wmma_bf16(a0, b[j], c0[j]);
            c1[j] = wmma_bf16(a1, b[j], c1[j]);
        }
    }
    // C/D layout: VGPR v, lanes 0-15 -> M=v, lanes 16-31 -> M=8+v, N=lane&15.
    float* E = energyp + (size_t)p * C_DIM * C_DIM;
    int n = lane & 15, mh = (lane >> 4) * 8;
    #pragma unroll
    for (int j = 0; j < 4; ++j) {
        #pragma unroll
        for (int v = 0; v < 8; ++v) {
            E[(size_t)(m0 + mh + v)      * C_DIM + (n0 + 16 * j + n)] = c0[j][v];
            E[(size_t)(m0 + 16 + mh + v) * C_DIM + (n0 + 16 * j + n)] = c1[j][v];
        }
    }
}

// --- Kernel 3: sum K-split partials, attn = exp(rowmin - e) / sum -----------
__global__ __launch_bounds__(256)
void tsam_softmax(const float* __restrict__ energyp, bf16* __restrict__ attn) {
    int row = blockIdx.x, tid = threadIdx.x;
    const size_t PS = (size_t)C_DIM * C_DIM;
    const float* e = energyp + (size_t)row * C_DIM;
    float v0 = 0.f, v1 = 0.f, v2 = 0.f, v3 = 0.f;
    #pragma unroll
    for (int p = 0; p < KSPLIT; ++p) {
        v0 += e[p * PS + tid];
        v1 += e[p * PS + tid + 256];
        v2 += e[p * PS + tid + 512];
        v3 += e[p * PS + tid + 768];
    }
    __shared__ float red[256];
    red[tid] = fminf(fminf(v0, v1), fminf(v2, v3));
    __syncthreads();
    for (int s = 128; s > 0; s >>= 1) {
        if (tid < s) red[tid] = fminf(red[tid], red[tid + s]);
        __syncthreads();
    }
    float rmin = red[0];
    __syncthreads();
    float e0 = __expf(rmin - v0), e1 = __expf(rmin - v1);
    float e2 = __expf(rmin - v2), e3 = __expf(rmin - v3);
    red[tid] = e0 + e1 + e2 + e3;
    __syncthreads();
    for (int s = 128; s > 0; s >>= 1) {
        if (tid < s) red[tid] += red[tid + s];
        __syncthreads();
    }
    float inv = 1.0f / red[0];
    bf16* a = attn + (size_t)row * C_DIM;
    a[tid]       = (bf16)(e0 * inv);
    a[tid + 256] = (bf16)(e1 * inv);
    a[tid + 512] = (bf16)(e2 * inv);
    a[tid + 768] = (bf16)(e3 * inv);
}

// --- Kernel 4: out = gamma * (attn @ q) + x  (B operand = qbT) --------------
__global__ __launch_bounds__(32)
void tsam_gemm2(const bf16* __restrict__ attn, const bf16* __restrict__ qbT,
                const float* __restrict__ x, const float* __restrict__ gamma,
                float* __restrict__ out) {
    int lane = threadIdx.x;
    int n0 = blockIdx.x * 64;   // HW columns
    int m0 = blockIdx.y * 32;   // channel rows
    v8f c0[4] = {{}, {}, {}, {}};
    v8f c1[4] = {{}, {}, {}, {}};
    #pragma unroll 2
    for (int k0 = 0; k0 < C_DIM; k0 += 32) {
        v16bf a0 = frag_a(attn, C_DIM, m0,      k0, lane);
        v16bf a1 = frag_a(attn, C_DIM, m0 + 16, k0, lane);
        v16bf b[4];
        #pragma unroll
        for (int j = 0; j < 4; ++j) b[j] = frag_b(qbT, C_DIM, n0 + 16 * j, k0, lane);
        #pragma unroll
        for (int j = 0; j < 4; ++j) {
            c0[j] = wmma_bf16(a0, b[j], c0[j]);
            c1[j] = wmma_bf16(a1, b[j], c1[j]);
        }
    }
    float g = gamma[0];
    int n = lane & 15, mh = (lane >> 4) * 8;
    #pragma unroll
    for (int j = 0; j < 4; ++j) {
        #pragma unroll
        for (int v = 0; v < 8; ++v) {
            size_t i0 = (size_t)(m0 + mh + v)      * HW + (n0 + 16 * j + n);
            size_t i1 = (size_t)(m0 + 16 + mh + v) * HW + (n0 + 16 * j + n);
            out[i0] = g * c0[j][v] + x[i0];
            out[i1] = g * c1[j][v] + x[i1];
        }
    }
}

// ---------------------------------------------------------------------------
extern "C" void kernel_launch(void* const* d_in, const int* in_sizes, int n_in,
                              void* d_out, int out_size, void* d_ws, size_t ws_size,
                              hipStream_t stream) {
    const float* x     = (const float*)d_in[0];
    const float* gamma = (const float*)d_in[1];
    float*       out   = (float*)d_out;

    // Workspace layout (82 MB total):
    //   qb     : bf16 [1024][16384]     @ 0      (32 MB)
    //   qbT    : bf16 [16384][1024]     @ 32 MB  (32 MB)
    //   energyp: f32  [4][1024][1024]   @ 64 MB  (16 MB, K-split partials)
    //   attn   : bf16 [1024][1024]      @ 80 MB  ( 2 MB)
    char* ws = (char*)d_ws;
    bf16*  qb      = (bf16*)(ws);
    bf16*  qbT     = (bf16*)(ws + (size_t)32 * 1024 * 1024);
    float* energyp = (float*)(ws + (size_t)64 * 1024 * 1024);
    bf16*  attn    = (bf16*)(ws + (size_t)80 * 1024 * 1024);

    tsam_pack<<<dim3(HW / 32, C_DIM / 32), dim3(32, 8), 0, stream>>>(x, qb, qbT);
    tsam_gemm1<<<dim3(C_DIM / 64, C_DIM / 32, KSPLIT), 32, 0, stream>>>(qb, energyp);
    tsam_softmax<<<C_DIM, 256, 0, stream>>>(energyp, attn);
    tsam_gemm2<<<dim3(HW / 64, C_DIM / 32), 32, 0, stream>>>(attn, qbT, x, gamma, out);
}
